// BiLSTM_CRF_43748536877066
// MI455X (gfx1250) — matrix operand order repaired and checked by reading, not compiled, not run on hardware
//
#include <hip/hip_runtime.h>
#include <hip/hip_bf16.h>
#include <math.h>

typedef __attribute__((ext_vector_type(16))) _Float16 v16h;
typedef __attribute__((ext_vector_type(8)))  _Float16 v8h;
typedef __attribute__((ext_vector_type(8)))  float    v8f;
typedef __attribute__((ext_vector_type(4)))  unsigned int u32x4;
typedef __attribute__((ext_vector_type(8)))  int      i32x8;
typedef __attribute__((ext_vector_type(4)))  int      i32x4;

#define EMB 100
#define EMBP 128     // padded embedding row (f16)
#define HH  100      // per-direction hidden
#define G4  400      // 4*H gates
#define KPAD 256     // concat [x(100 pad128) | h(100 pad128)]
#define Bsz 512
#define Lsz 512
#define KTAG 20
#define START_TAG 18
#define END_TAG 19
#define NEGV -1e6f
#define SLAB 16      // batch rows per workgroup in the recurrence

// ---------------- WMMA fragment helpers (contiguous k = 16h + s, same order for A and B) ------

__device__ __forceinline__ v8f wmma16(v16h a, v16h b, v8f c) {
  return __builtin_amdgcn_wmma_f32_16x16x32_f16(
      /*neg_a=*/false, a, /*neg_b=*/false, b,
      /*c_mod=*/(short)0, c, /*reuse_a=*/false, /*reuse_b=*/false);
}

// A: 16x32 f16. lane<16: row=lane, K = k0..k0+15
//               lane>=16: row=lane-16, K = k0+16..k0+31
__device__ __forceinline__ v16h load_a_frag(const _Float16* base, int stride,
                                            int lane, int k0) {
  const _Float16* r = base + (lane & 15) * stride + k0 + ((lane & 16) ? 16 : 0);
  return *(const v16h*)r;
}

// B: 32x16 f16 stored as W[n][k] (row-major weights, gemm uses W^T).
// lane<16: col=lane, K=k0..k0+15 ; lane>=16: col=lane-16, K=k0+16..k0+31
__device__ __forceinline__ v16h load_b_frag(const _Float16* w, int stride,
                                            int lane, int n0, int k0) {
  const _Float16* r = w + (n0 + (lane & 15)) * stride + k0 + ((lane & 16) ? 16 : 0);
  return *(const v16h*)r;   // 16 contiguous halfs, 32B aligned
}

__device__ __forceinline__ float sigm(float x) { return 1.f / (1.f + __expf(-x)); }

// ---------------- Tensor Data Mover: 2D tile global -> LDS ------------------
// Descriptor per CDNA5 ISA 8.3/8.4. tensor == tile (no OOB), data_size = 2B.
// Optional LDS padding: pad_interval/amount codes per D# group 1.
__device__ __forceinline__ void tdm_load_2d(unsigned lds_off, const void* gptr,
                                            unsigned d0, unsigned d1,
                                            unsigned long long stride0_elems,
                                            bool pad, unsigned pad_icode,
                                            unsigned pad_acode) {
  unsigned long long ga = (unsigned long long)(uintptr_t)gptr;
  u32x4 g0;
  g0[0] = 1u;                                        // count=1, user mode
  g0[1] = lds_off;                                   // lds_addr [63:32]
  g0[2] = (unsigned)(ga & 0xFFFFFFFFu);              // global_addr [95:64]
  g0[3] = (unsigned)((ga >> 32) & 0x01FFFFFFu)       // global_addr [120:96]
          | (2u << 30);                              // type=2 ("image")
  i32x8 g1;
  g1[0] = (int)((1u << 16)                           // data_size = 2 bytes
                | (pad ? (1u << 20) : 0u)            // pad_enable
                | (pad_icode << 22)                  // pad_interval
                | (pad_acode << 25));                // pad_amount
  g1[1] = (int)((d0 & 0xFFFFu) << 16);               // tensor_dim0[15:0]
  g1[2] = (int)(((d0 >> 16) & 0xFFFFu) | ((d1 & 0xFFFFu) << 16)); // dim0 hi | dim1 lo
  g1[3] = (int)(((d1 >> 16) & 0xFFFFu) | ((d0 & 0xFFFFu) << 16)); // dim1 hi | tile_dim0
  g1[4] = (int)(d1 & 0xFFFFu);                       // tile_dim1 (tile_dim2 = 0)
  g1[5] = (int)(stride0_elems & 0xFFFFFFFFu);        // tensor_dim0_stride lo
  g1[6] = (int)((stride0_elems >> 32) & 0xFFFFu);    // stride hi (dim1_stride = 0)
  g1[7] = 0;
  i32x4 z4 = {0, 0, 0, 0};
#if __clang_major__ >= 23
  i32x8 z8 = {0, 0, 0, 0, 0, 0, 0, 0};
  __builtin_amdgcn_tensor_load_to_lds(g0, g1, z4, z4, z8, 0);
#else
  __builtin_amdgcn_tensor_load_to_lds(g0, g1, z4, z4, 0);
#endif
}

// ---------------- prep kernels ----------------------------------------------

// one block per (b,t): gather embedding rows, zero-padded to 128 halfs,
// plus reversed-within-length gather (packed-BiLSTM emulation)
__global__ void gather_kernel(const int* __restrict__ X,
                              const int* __restrict__ lengths,
                              const float* __restrict__ emb,
                              _Float16* __restrict__ xe,
                              _Float16* __restrict__ xr) {
  int rowg = blockIdx.x;              // b*L + t
  int b = rowg >> 9, t = rowg & 511;
  int len = lengths[b];
  int rv = (t < len) ? (len - 1 - t) : t;
  int xi  = X[rowg];
  int xir = X[(b << 9) + rv];
  int j = threadIdx.x;                // blockDim = 128
  _Float16 ve = (_Float16)0.f, vr = (_Float16)0.f;
  if (j < EMB) {
    ve = (_Float16)emb[(size_t)xi  * EMB + j];
    vr = (_Float16)emb[(size_t)xir * EMB + j];
  }
  xe[(size_t)rowg * EMBP + j] = ve;
  xr[(size_t)rowg * EMBP + j] = vr;
}

// pack weights into zero-padded f16 layouts
__global__ void pack_kernel(const float* __restrict__ Wih_f, const float* __restrict__ Whh_f,
                            const float* __restrict__ Wih_b, const float* __restrict__ Whh_b,
                            const float* __restrict__ W1,    const float* __restrict__ W2,
                            _Float16* __restrict__ wcf, _Float16* __restrict__ wcb,
                            _Float16* __restrict__ w1c, _Float16* __restrict__ w2c) {
  int tid = blockIdx.x * blockDim.x + threadIdx.x;
  int nthr = gridDim.x * blockDim.x;
  for (int i = tid; i < G4 * KPAD; i += nthr) {          // [Wih | Whh] 400x256
    int n = i >> 8, k = i & 255;
    float vf = 0.f, vb = 0.f;
    if (k < 100)                  { vf = Wih_f[n * 100 + k];         vb = Wih_b[n * 100 + k]; }
    else if (k >= 128 && k < 228) { vf = Whh_f[n * 100 + (k - 128)]; vb = Whh_b[n * 100 + (k - 128)]; }
    wcf[i] = (_Float16)vf; wcb[i] = (_Float16)vb;
  }
  for (int i = tid; i < 64 * 224; i += nthr) {           // W1: 50x200 -> 64x224
    int n = i / 224, k = i % 224;
    float v = (n < 50 && k < 200) ? W1[n * 200 + k] : 0.f;
    w1c[i] = (_Float16)v;
  }
  for (int i = tid; i < 32 * 64; i += nthr) {            // W2: 20x50 -> 32x64
    int n = i >> 6, k = i & 63;
    float v = (n < 20 && k < 50) ? W2[n * 50 + k] : 0.f;
    w2c[i] = (_Float16)v;
  }
}

// ---------------- recurrent LSTM (one WG = 16 batch rows, loops all t) ------
// grid = (32, 2): x = batch slab, y = direction. block = 256 (8 waves).
__global__ void lstm_kernel(const _Float16* __restrict__ xe,
                            const _Float16* __restrict__ xr,
                            const _Float16* __restrict__ wcf,
                            const _Float16* __restrict__ wcb,
                            const float* __restrict__ b_f,
                            const float* __restrict__ b_b,
                            const float* __restrict__ h0,
                            const float* __restrict__ c0,
                            _Float16* __restrict__ hf_out,
                            _Float16* __restrict__ hb_out) {
  extern __shared__ char smem[];
  _Float16* Wl = (_Float16*)smem;              // 400*256 f16 = 204800 B
  _Float16* Al = Wl + G4 * KPAD;               // 16*256  f16 = 8192 B
  float*    Gl = (float*)(Al + SLAB * KPAD);   // 16*400  f32 = 25600 B
  float*    Hl = Gl + SLAB * G4;               // 16*100  f32 = 6400 B
  float*    Cl = Hl + SLAB * HH;               // 16*100  f32 = 6400 B

  int tid = threadIdx.x, lane = tid & 31, wave = tid >> 5;
  int dir = blockIdx.y;
  int b0  = blockIdx.x * SLAB;
  const _Float16* xsrc = dir ? xr : xe;
  const _Float16* wsrc = dir ? wcb : wcf;
  const float* bias = dir ? b_b : b_f;
  _Float16* hout = dir ? hb_out : hf_out;

  // TDM: weights 400x256 f16 -> LDS (contiguous, no pad); wave 0 issues once.
  if (wave == 0) {
    tdm_load_2d((unsigned)(uintptr_t)(void*)Wl, wsrc,
                /*d0=*/KPAD, /*d1=*/G4, /*stride0=*/KPAD,
                /*pad=*/false, 0, 0);
  }
  for (int i = tid; i < SLAB * HH; i += 256) {           // h0/c0 -> LDS
    int r = i / HH, j = i % HH;
    Hl[i] = h0[(size_t)dir * Bsz * HH + (b0 + r) * HH + j];
    Cl[i] = c0[(size_t)dir * Bsz * HH + (b0 + r) * HH + j];
  }
  if (wave == 0) __builtin_amdgcn_s_wait_tensorcnt(0);
  __syncthreads();

  unsigned al_off = (unsigned)(uintptr_t)(void*)Al;
  const _Float16* xbase = xsrc + (size_t)b0 * Lsz * EMBP;

  for (int t = 0; t < Lsz; ++t) {
    // TDM: x_t slab 16 rows x 128 halfs -> Al at 256-half pitch.
    // pad_interval code 5 = 64 DWORDs (=128 halfs), pad_amount code 63 =
    // 64 DWORDs -> skips the h-half of each 256-half row.
    if (wave == 0) {
      tdm_load_2d(al_off, xbase + (size_t)t * EMBP,
                  /*d0=*/EMBP, /*d1=*/SLAB,
                  /*stride0=*/(unsigned long long)Lsz * EMBP,
                  /*pad=*/true, /*icode=*/5, /*acode=*/63);
    }
    // VALU fills the h-half (k = 128..255) from Hl
    for (int i = tid; i < SLAB * EMBP; i += 256) {
      int r = i >> 7, k = i & 127;
      Al[r * KPAD + EMBP + k] = (k < HH) ? (_Float16)Hl[r * HH + k]
                                         : (_Float16)0.f;
    }
    if (wave == 0) __builtin_amdgcn_s_wait_tensorcnt(0);
    __syncthreads();

    // gates = A @ Wcat^T : 1 Mtile x 25 Ntiles, 8 K-chunks of 32
    for (int nt = wave; nt < 25; nt += 8) {
      v8f acc = {};
#pragma unroll
      for (int kc = 0; kc < 8; ++kc) {
        v16h a  = load_a_frag(Al, KPAD, lane, kc * 32);
        v16h bb = load_b_frag(Wl, KPAD, lane, nt * 16, kc * 32);
        acc = wmma16(a, bb, acc);
      }
      int col   = nt * 16 + (lane & 15);
      int rbase = (lane & 16) ? 8 : 0;
#pragma unroll
      for (int v = 0; v < 8; ++v) Gl[(rbase + v) * G4 + col] = acc[v];
    }
    __syncthreads();

    // activations: gate order i, f, g, o (chunks of 100)
    for (int i = tid; i < SLAB * HH; i += 256) {
      int r = i / HH, j = i % HH;
      float gi = Gl[r * G4 + j]        + bias[j];
      float gf = Gl[r * G4 + 100 + j]  + bias[100 + j];
      float gg = Gl[r * G4 + 200 + j]  + bias[200 + j];
      float go = Gl[r * G4 + 300 + j]  + bias[300 + j];
      float cn = sigm(gf) * Cl[i] + sigm(gi) * tanhf(gg);
      float hn = sigm(go) * tanhf(cn);
      Cl[i] = cn; Hl[i] = hn;
      hout[((size_t)(b0 + r) * Lsz + t) * HH + j] = (_Float16)hn;
    }
    __syncthreads();
  }
}

// ---------------- MLP: probs = relu(enc@W1^T+b1)@W2^T+b2 --------------------
// block = 256 (8 waves), 128 rows/block (1 Mtile per wave). grid = 2048.
__global__ void mlp_kernel(const _Float16* __restrict__ hf,
                           const _Float16* __restrict__ hb,
                           const int* __restrict__ lengths,
                           const _Float16* __restrict__ w1c,
                           const _Float16* __restrict__ w2c,
                           const float* __restrict__ b1,
                           const float* __restrict__ b2,
                           float* __restrict__ probs) {
  extern __shared__ char smem[];
  _Float16* Enc = (_Float16*)smem;          // 128*224 f16 = 57344 B
  _Float16* W1l = Enc + 128 * 224;          // 64*224  f16 = 28672 B
  _Float16* Hm  = W1l + 64 * 224;           // 128*64  f16 = 16384 B
  _Float16* W2l = Hm + 128 * 64;            // 32*64   f16 = 4096 B
  float*    b1l = (float*)(W2l + 32 * 64);  // 64 f32
  float*    b2l = b1l + 64;                 // 32 f32

  int tid = threadIdx.x, lane = tid & 31, wave = tid >> 5;
  int row0 = blockIdx.x * 128;

  for (int i = tid; i < 64 * 224 / 8; i += 256) ((int4*)W1l)[i] = ((const int4*)w1c)[i];
  for (int i = tid; i < 32 * 64 / 8; i += 256)  ((int4*)W2l)[i] = ((const int4*)w2c)[i];
  if (tid < 64) b1l[tid] = (tid < 50) ? b1[tid] : 0.f;
  if (tid < 32) b2l[tid] = (tid < 20) ? b2[tid] : 0.f;

  // stage enc = [hf | hb_unreversed], zeroed past length, K padded to 224
  for (int i = tid; i < 128 * 224; i += 256) {
    int r = i / 224, k = i % 224;
    int rowg = row0 + r;
    int b = rowg >> 9, t = rowg & 511;
    int len = lengths[b];
    _Float16 v = (_Float16)0.f;
    if (t < len) {
      if (k < 100) v = hf[(size_t)rowg * HH + k];
      else if (k < 200) {
        int rv = len - 1 - t;
        v = hb[((size_t)(b << 9) + rv) * HH + (k - 100)];
      }
    }
    Enc[i] = v;
  }
  __syncthreads();

  // layer 1: K=224 (7 chunks), N=64 (4 Ntiles)
  {
    v8f acc[4]; v8f z = {};
#pragma unroll
    for (int n = 0; n < 4; ++n) acc[n] = z;
    for (int kc = 0; kc < 7; ++kc) {
      v16h a = load_a_frag(Enc + wave * 16 * 224, 224, lane, kc * 32);
#pragma unroll
      for (int n = 0; n < 4; ++n) {
        v16h bb = load_b_frag(W1l, 224, lane, n * 16, kc * 32);
        acc[n] = wmma16(a, bb, acc[n]);
      }
    }
    int rbase = wave * 16 + ((lane & 16) ? 8 : 0);
#pragma unroll
    for (int n = 0; n < 4; ++n) {
      int col = n * 16 + (lane & 15);
      float bb1 = b1l[col];
#pragma unroll
      for (int v = 0; v < 8; ++v) {
        float x = acc[n][v] + bb1;
        Hm[(rbase + v) * 64 + col] = (_Float16)(x > 0.f ? x : 0.f);
      }
    }
  }
  __syncthreads();

  // layer 2: K=64 (2 chunks), N=32 (2 Ntiles), valid cols < 20
  {
    v8f acc[2]; v8f z = {}; acc[0] = z; acc[1] = z;
    for (int kc = 0; kc < 2; ++kc) {
      v16h a = load_a_frag(Hm + wave * 16 * 64, 64, lane, kc * 32);
#pragma unroll
      for (int n = 0; n < 2; ++n) {
        v16h bb = load_b_frag(W2l, 64, lane, n * 16, kc * 32);
        acc[n] = wmma16(a, bb, acc[n]);
      }
    }
    int rbase = wave * 16 + ((lane & 16) ? 8 : 0);
#pragma unroll
    for (int n = 0; n < 2; ++n) {
      int col = n * 16 + (lane & 15);
      if (col < KTAG) {
        float bb2 = b2l[col];
#pragma unroll
        for (int v = 0; v < 8; ++v) {
          int rowg = row0 + rbase + v;
          probs[(size_t)rowg * KTAG + col] = acc[n][v] + bb2;
        }
      }
    }
  }
}

// ---------------- Viterbi: one wave32 per batch row -------------------------
// block = 128 (4 waves), grid = 128.
__global__ void viterbi_kernel(const float* __restrict__ probs,
                               const int* __restrict__ lengths,
                               const float* __restrict__ trans,
                               unsigned char* __restrict__ bp,
                               float* __restrict__ out) {
  __shared__ float transl[KTAG * KTAG];
  __shared__ float alpha[4][32];
  __shared__ float finall[4][32];
  int tid = threadIdx.x, lane = tid & 31, wave = tid >> 5;
  for (int i = tid; i < KTAG * KTAG; i += 128) transl[i] = trans[i];
  int b = blockIdx.x * 4 + wave;
  int len = lengths[b];
  alpha[wave][lane] = (lane == START_TAG) ? 0.f : NEGV;
  __syncthreads();

  int j = lane;
  int jc = (j < KTAG) ? j : 0;
  for (int t = 0; t < Lsz; ++t) {
    float m = -3.4e38f; int am = 0;
#pragma unroll 4
    for (int i = 0; i < KTAG; ++i) {
      float v = alpha[wave][i] + transl[i * KTAG + jc];
      if (v > m) { m = v; am = i; }
    }
    if (j < KTAG) {
      bp[((size_t)b * Lsz + t) * KTAG + j] = (unsigned char)am;
      if (t < len) {
        float p = probs[((size_t)b * Lsz + t) * KTAG + j];
        alpha[wave][j] = m + p;      // wave-lockstep: reads above precede write
      }
    }
  }
  finall[wave][lane] = (j < KTAG) ? (alpha[wave][j] + transl[j * KTAG + END_TAG])
                                  : -3.4e38f;
  if (lane == 0) {
    float best = finall[wave][0]; int cur = 0;
    for (int i = 1; i < KTAG; ++i)
      if (finall[wave][i] > best) { best = finall[wave][i]; cur = i; }
    out[b] = best;
    for (int t = Lsz - 1; t >= 0; --t) {
      if (t < len) {
        out[Bsz + (size_t)b * Lsz + t] = (float)cur;
        cur = bp[((size_t)b * Lsz + t) * KTAG + cur];
      } else {
        out[Bsz + (size_t)b * Lsz + t] = -1.f;
      }
    }
  }
}

// ---------------- host launcher ---------------------------------------------

extern "C" void kernel_launch(void* const* d_in, const int* in_sizes, int n_in,
                              void* d_out, int out_size, void* d_ws, size_t ws_size,
                              hipStream_t stream) {
  const int*   X       = (const int*)d_in[0];
  const int*   lengths = (const int*)d_in[1];
  const float* emb     = (const float*)d_in[2];
  const float* Wih_f   = (const float*)d_in[3];
  const float* Whh_f   = (const float*)d_in[4];
  const float* b_f     = (const float*)d_in[5];
  const float* Wih_b   = (const float*)d_in[6];
  const float* Whh_b   = (const float*)d_in[7];
  const float* b_b     = (const float*)d_in[8];
  const float* h0      = (const float*)d_in[9];
  const float* c0      = (const float*)d_in[10];
  const float* W1      = (const float*)d_in[11];
  const float* b1      = (const float*)d_in[12];
  const float* W2      = (const float*)d_in[13];
  const float* b2      = (const float*)d_in[14];
  const float* trans   = (const float*)d_in[15];
  float* out = (float*)d_out;

  char* ws = (char*)d_ws;
  size_t off = 0;
  auto alloc = [&](size_t bytes) -> void* {
    void* p = ws + off;
    off = (off + bytes + 255) & ~(size_t)255;
    return p;
  };
  const size_t nBL = (size_t)Bsz * Lsz;
  _Float16* xe  = (_Float16*)alloc(nBL * EMBP * 2);
  _Float16* xr  = (_Float16*)alloc(nBL * EMBP * 2);
  _Float16* hf  = (_Float16*)alloc(nBL * HH * 2);
  _Float16* hb  = (_Float16*)alloc(nBL * HH * 2);
  float*    probs = (float*)alloc(nBL * KTAG * 4);
  unsigned char* bp = (unsigned char*)alloc(nBL * KTAG);
  _Float16* wcf = (_Float16*)alloc((size_t)G4 * KPAD * 2);
  _Float16* wcb = (_Float16*)alloc((size_t)G4 * KPAD * 2);
  _Float16* w1c = (_Float16*)alloc((size_t)64 * 224 * 2);
  _Float16* w2c = (_Float16*)alloc((size_t)32 * 64 * 2);

  const uint32_t lstmSmem = G4 * KPAD * 2 + SLAB * KPAD * 2 + SLAB * G4 * 4
                          + 2 * SLAB * HH * 4;
  const uint32_t mlpSmem  = 128 * 224 * 2 + 64 * 224 * 2 + 128 * 64 * 2
                          + 32 * 64 * 2 + 64 * 4 + 32 * 4;
  hipFuncSetAttribute((const void*)lstm_kernel,
                      hipFuncAttributeMaxDynamicSharedMemorySize, (int)lstmSmem);
  hipFuncSetAttribute((const void*)mlp_kernel,
                      hipFuncAttributeMaxDynamicSharedMemorySize, (int)mlpSmem);

  gather_kernel<<<Bsz * Lsz, 128, 0, stream>>>(X, lengths, emb, xe, xr);
  pack_kernel<<<64, 256, 0, stream>>>(Wih_f, Whh_f, Wih_b, Whh_b, W1, W2,
                                      wcf, wcb, w1c, w2c);
  lstm_kernel<<<dim3(Bsz / SLAB, 2), 256, lstmSmem, stream>>>(
      xe, xr, wcf, wcb, b_f, b_b, h0, c0, hf, hb);
  mlp_kernel<<<(Bsz * Lsz) / 128, 256, mlpSmem, stream>>>(
      hf, hb, lengths, w1c, w2c, b1, b2, probs);
  viterbi_kernel<<<Bsz / 4, 128, 0, stream>>>(probs, lengths, trans, bp, out);
}
